// AdaptiveMixtureOfExperts_85607288143942
// MI455X (gfx1250) — matrix operand order repaired and checked by reading, not compiled, not run on hardware
//
#include <hip/hip_runtime.h>
#include <hip/hip_bf16.h>

// ---------------------------------------------------------------------------
// MoE (top-2 of 8 experts, SwiGLU) forward for MI455X (gfx1250, wave32).
// bf16x2-split WMMA (hi+lo planes, 3 products) for fp32-level accuracy.
// Weights/activations pre-split into bf16 hi/lo planes so GEMM K-loops stream
// tiles with GLOBAL_LOAD_ASYNC_TO_LDS_B128 (ASYNCcnt) into double-buffered LDS.
// ---------------------------------------------------------------------------

#define T_TOK 4096      // B*S tokens
#define DM    1024      // d_model
#define DFF   2048      // d_ff
#define H2    4096      // 2*d_ff
#define NE    8         // experts
#define LDT   40        // LDS K-stride in halfs (32 + 8 pad; 80B, 16B multiple)

typedef __attribute__((ext_vector_type(16))) __bf16 v16bf;
typedef __attribute__((ext_vector_type(8)))  float  v8f;

union Frag16 { v16bf v; uint4 q[2]; };

__device__ __forceinline__ unsigned short f2bf(float f) {
    unsigned int b = __builtin_bit_cast(unsigned int, f);
    b += 0x7FFFu + ((b >> 16) & 1u);
    return (unsigned short)(b >> 16);
}
__device__ __forceinline__ float bf2f(unsigned short h) {
    unsigned int b = ((unsigned int)h) << 16;
    return __builtin_bit_cast(float, b);
}
__device__ __forceinline__ void split_bf16(float x, unsigned short* hi, unsigned short* lo) {
    unsigned short h = f2bf(x);
    *hi = h;
    *lo = f2bf(x - bf2f(h));
}

__device__ __forceinline__ v8f wmma_bf16(v16bf a, v16bf b, v8f c) {
    return __builtin_amdgcn_wmma_f32_16x16x32_bf16(false, a, false, b, (short)0, c, false, false);
}

// raw 32-bit LDS offset of a __shared__ object (addrspace(3) pointer value)
__device__ __forceinline__ unsigned lds_lo32(const void* p) {
    return (unsigned)(unsigned long long)
        (__attribute__((address_space(3))) const unsigned char*)p;
}

// one 16B async copy global->LDS (per-lane addresses), tracked by ASYNCcnt
__device__ __forceinline__ void async_cp16(unsigned lds_addr, const void* gptr) {
    unsigned long long ga = (unsigned long long)gptr;
    asm volatile("global_load_async_to_lds_b128 %0, %1, off"
                 :: "v"(lds_addr), "v"(ga) : "memory");
}
#define WAIT_ASYNC_5() asm volatile("s_wait_asynccnt 0x5" ::: "memory")
#define WAIT_ASYNC_0() asm volatile("s_wait_asynccnt 0x0" ::: "memory")

// A-fragment (16x32 bf16): lane l: M=l%16, K = t + 8*((l/16)+(t/8))
__device__ __forceinline__ v16bf load_frag_a(const unsigned short* rowbase, int hs) {
    Frag16 f;
    f.q[0] = *(const uint4*)(rowbase + 8 * hs);
    f.q[1] = *(const uint4*)(rowbase + 16 + 8 * hs);
    return f.v;
}
// B-fragment (32x16 bf16), K-contiguous per column: lane l: N=l%16, K = t + 8*((l/16)+(t/8))
__device__ __forceinline__ v16bf load_frag_b(const unsigned short* colbase, int hs) {
    Frag16 f;
    f.q[0] = *(const uint4*)(colbase + 8 * hs);
    f.q[1] = *(const uint4*)(colbase + 16 + 8 * hs);
    return f.v;
}

// ---------------------------------------------------------------------------
// split fp32 vector -> bf16 hi/lo planes (for x)
// ---------------------------------------------------------------------------
__global__ __launch_bounds__(256) void split_plane(
    const float* __restrict__ src, unsigned short* __restrict__ hi,
    unsigned short* __restrict__ lo, int n)
{
    int i = blockIdx.x * 256 + threadIdx.x;
    if (i < n) split_bf16(src[i], &hi[i], &lo[i]);
}

// ---------------------------------------------------------------------------
// per-expert transpose + split: in[e][k][n] (fp32) -> planes[e][n][k] (bf16)
// LDS-tiled 32x32 so both global sides are coalesced.
// ---------------------------------------------------------------------------
__global__ __launch_bounds__(256) void transpose_split(
    const float* __restrict__ in, unsigned short* __restrict__ hi,
    unsigned short* __restrict__ lo, int K, int N)
{
    const int e  = blockIdx.z;
    const int k0 = blockIdx.y * 32;
    const int n0 = blockIdx.x * 32;
    __shared__ float t[32][33];
    const float* src = in + (size_t)e * K * N;
    const int tid = threadIdx.x;
#pragma unroll
    for (int i = tid; i < 1024; i += 256) {
        int kk = i >> 5, nn = i & 31;
        t[kk][nn] = src[(size_t)(k0 + kk) * N + n0 + nn];
    }
    __syncthreads();
#pragma unroll
    for (int i = tid; i < 1024; i += 256) {
        int nn = i >> 5, kk = i & 31;         // kk fastest -> coalesced stores
        size_t o = (size_t)e * N * K + (size_t)(n0 + nn) * K + k0 + kk;
        split_bf16(t[kk][nn], &hi[o], &lo[o]);
    }
}

// ---------------------------------------------------------------------------
// router: one wave32 per token, fp32 dots, exact top-2, softmax(2), append.
// ---------------------------------------------------------------------------
__global__ __launch_bounds__(256) void moe_router(
    const float* __restrict__ x, const float* __restrict__ Wr,
    const float* __restrict__ temp,
    int* __restrict__ cnt, int* __restrict__ tok_of,
    int* __restrict__ slot_of, float* __restrict__ wgt_of)
{
    const int tid  = threadIdx.x;
    const int lane = tid & 31;
    const int t    = blockIdx.x * 8 + (tid >> 5);
    if (t >= T_TOK) return;

    float acc[NE];
#pragma unroll
    for (int e = 0; e < NE; ++e) acc[e] = 0.f;

    const float* xr = x + (size_t)t * DM;
    for (int d = lane; d < DM; d += 32) {
        float xv = xr[d];
        const float4* wr = (const float4*)(Wr + (size_t)d * NE);
        float4 w0 = wr[0], w1 = wr[1];
        acc[0] += xv * w0.x; acc[1] += xv * w0.y;
        acc[2] += xv * w0.z; acc[3] += xv * w0.w;
        acc[4] += xv * w1.x; acc[5] += xv * w1.y;
        acc[6] += xv * w1.z; acc[7] += xv * w1.w;
    }
#pragma unroll
    for (int e = 0; e < NE; ++e) {
#pragma unroll
        for (int off = 16; off >= 1; off >>= 1)
            acc[e] += __shfl_xor(acc[e], off, 32);
    }

    if (lane == 0) {
        const float inv_t = 1.0f / temp[0];
        float l0 = -3.4e38f, l1 = -3.4e38f;
        int e0 = 0, e1 = 0;
#pragma unroll
        for (int e = 0; e < NE; ++e) {
            float v = acc[e] * inv_t;
            if (v > l0)      { l1 = l0; e1 = e0; l0 = v; e0 = e; }
            else if (v > l1) { l1 = v;  e1 = e; }
        }
        float p0 = 1.0f / (1.0f + expf(l1 - l0));
        float p1 = 1.0f - p0;

        int pos0 = atomicAdd(&cnt[e0], 1);
        tok_of [e0 * T_TOK + pos0] = t;
        slot_of[e0 * T_TOK + pos0] = 2 * t + 0;
        wgt_of [e0 * T_TOK + pos0] = p0;
        int pos1 = atomicAdd(&cnt[e1], 1);
        tok_of [e1 * T_TOK + pos1] = t;
        slot_of[e1 * T_TOK + pos1] = 2 * t + 1;
        wgt_of [e1 * T_TOK + pos1] = p1;
    }
}

// ---------------------------------------------------------------------------
// GEMM1 + SwiGLU. Tile: 32 rows x (64 'a' + 64 'g') cols, K-tile 32,
// double-buffered LDS fed by async copies. 8 waves: (mf = w&1, np = w>>1).
// Writes u as bf16 hi/lo planes (K-contiguous for GEMM2's async A-path).
// ---------------------------------------------------------------------------
__global__ __launch_bounds__(256) void moe_gemm1(
    const unsigned short* __restrict__ xhi, const unsigned short* __restrict__ xlo,
    const unsigned short* __restrict__ W1t_hi, const unsigned short* __restrict__ W1t_lo,
    const float* __restrict__ b1,
    const int* __restrict__ cnt, const int* __restrict__ tok_of,
    const int* __restrict__ slot_of,
    unsigned short* __restrict__ u_hi, unsigned short* __restrict__ u_lo)
{
    const int e  = blockIdx.z;
    const int m0 = blockIdx.y * 32;
    const int n0 = blockIdx.x * 64;
    const int c  = cnt[e];
    if (m0 >= c) return;

    __shared__ __align__(16) unsigned short Ah[2][32 * LDT];
    __shared__ __align__(16) unsigned short Al[2][32 * LDT];
    __shared__ __align__(16) unsigned short Bh[2][128 * LDT];
    __shared__ __align__(16) unsigned short Bl[2][128 * LDT];

    const int tid  = threadIdx.x;
    const int lane = tid & 31;
    const int wv   = tid >> 5;
    const int hs   = lane >> 4;
    const int lid  = lane & 15;

    // ---- per-thread async-copy descriptors (hoisted out of K loop) ----
    // A: 256 chunks = 32 rows x 4 x {hi,lo}; 1 per thread
    const int ap = tid >> 7, at = tid & 127, ar = at >> 2, aq = at & 3;
    int arow = m0 + ar; if (arow > c - 1) arow = c - 1;
    const int atok = tok_of[e * T_TOK + arow];
    const unsigned short* asrc = (ap ? xlo : xhi) + (size_t)atok * DM + aq * 8;
    unsigned alds[2];
    alds[0] = lds_lo32(&(ap ? Al : Ah)[0][ar * LDT + aq * 8]);
    alds[1] = lds_lo32(&(ap ? Al : Ah)[1][ar * LDT + aq * 8]);

    // B: 1024 chunks = 128 cols x 4 x {hi,lo}; 4 per thread
    const unsigned short* bsrc[4];
    unsigned blds[4][2];
#pragma unroll
    for (int s = 0; s < 4; ++s) {
        int cid = tid + s * 256;
        int pb = cid >> 9, t3 = cid & 511, cc = t3 >> 2, qb = t3 & 3;
        int gcol = (cc < 64) ? (n0 + cc) : (DFF + n0 + (cc - 64));
        bsrc[s] = (pb ? W1t_lo : W1t_hi) + (size_t)(e * H2 + gcol) * DM + qb * 8;
        blds[s][0] = lds_lo32(&(pb ? Bl : Bh)[0][cc * LDT + qb * 8]);
        blds[s][1] = lds_lo32(&(pb ? Bl : Bh)[1][cc * LDT + qb * 8]);
    }

    const int mf = wv & 1, np = wv >> 1;
    v8f acc_a = {}, acc_g = {};

    // prologue: fill buffer 0
    async_cp16(alds[0], asrc);
#pragma unroll
    for (int s = 0; s < 4; ++s) async_cp16(blds[s][0], bsrc[s]);

    const int NT = DM / 32;
    for (int kt = 0; kt < NT; ++kt) {
        const int buf = kt & 1;
        if (kt + 1 < NT) {
            const int kb = (kt + 1) * 32;
            async_cp16(alds[buf ^ 1], asrc + kb);
#pragma unroll
            for (int s = 0; s < 4; ++s) async_cp16(blds[s][buf ^ 1], bsrc[s] + kb);
            WAIT_ASYNC_5();
        } else {
            WAIT_ASYNC_0();
        }
        __syncthreads();

        const unsigned short* Ahb = Ah[buf];
        const unsigned short* Alb = Al[buf];
        const unsigned short* Bhb = Bh[buf];
        const unsigned short* Blb = Bl[buf];

        v16bf ahf = load_frag_a(&Ahb[(mf * 16 + lid) * LDT], hs);
        v16bf alf = load_frag_a(&Alb[(mf * 16 + lid) * LDT], hs);
        int ca = np * 16 + lid;
        int cg = 64 + np * 16 + lid;
        v16bf bah = load_frag_b(&Bhb[ca * LDT], hs);
        v16bf bal = load_frag_b(&Blb[ca * LDT], hs);
        v16bf bgh = load_frag_b(&Bhb[cg * LDT], hs);
        v16bf bgl = load_frag_b(&Blb[cg * LDT], hs);

        acc_a = wmma_bf16(alf, bah, acc_a);
        acc_a = wmma_bf16(ahf, bal, acc_a);
        acc_a = wmma_bf16(ahf, bah, acc_a);
        acc_g = wmma_bf16(alf, bgh, acc_g);
        acc_g = wmma_bf16(ahf, bgl, acc_g);
        acc_g = wmma_bf16(ahf, bgh, acc_g);
        __syncthreads();
    }

    // epilogue: bias + SwiGLU, write u hi/lo planes
    const int ncol = n0 + np * 16 + lid;
    const float ba = b1[e * H2 + ncol];
    const float bg = b1[e * H2 + DFF + ncol];
#pragma unroll
    for (int j = 0; j < 8; ++j) {
        int grow = m0 + mf * 16 + j + 8 * hs;
        if (grow < c) {
            float a = acc_a[j] + ba;
            float g = acc_g[j] + bg;
            float val = a * (g / (1.0f + expf(-g)));
            int slot = slot_of[e * T_TOK + grow];
            unsigned short h, l;
            split_bf16(val, &h, &l);
            u_hi[(size_t)slot * DFF + ncol] = h;
            u_lo[(size_t)slot * DFF + ncol] = l;
        }
    }
}

// ---------------------------------------------------------------------------
// GEMM2. Tile: 32 rows x 128 cols, K = 2048, async double-buffered LDS.
// 8 waves: (mf = w&1, nh = w>>1), 2 col-frags per wave.
// ---------------------------------------------------------------------------
__global__ __launch_bounds__(256) void moe_gemm2(
    const unsigned short* __restrict__ u_hi, const unsigned short* __restrict__ u_lo,
    const unsigned short* __restrict__ W2t_hi, const unsigned short* __restrict__ W2t_lo,
    const float* __restrict__ b2,
    const int* __restrict__ cnt, const int* __restrict__ slot_of,
    const float* __restrict__ wgt_of, float* __restrict__ yslot)
{
    const int e  = blockIdx.z;
    const int m0 = blockIdx.y * 32;
    const int n0 = blockIdx.x * 128;
    const int c  = cnt[e];
    if (m0 >= c) return;

    __shared__ __align__(16) unsigned short Ah[2][32 * LDT];
    __shared__ __align__(16) unsigned short Al[2][32 * LDT];
    __shared__ __align__(16) unsigned short Bh[2][128 * LDT];
    __shared__ __align__(16) unsigned short Bl[2][128 * LDT];

    const int tid  = threadIdx.x;
    const int lane = tid & 31;
    const int wv   = tid >> 5;
    const int hs   = lane >> 4;
    const int lid  = lane & 15;

    // A: 1 chunk/thread from u planes (rows gathered by slot)
    const int ap = tid >> 7, at = tid & 127, ar = at >> 2, aq = at & 3;
    int arow = m0 + ar; if (arow > c - 1) arow = c - 1;
    const int aslot = slot_of[e * T_TOK + arow];
    const unsigned short* asrc = (ap ? u_lo : u_hi) + (size_t)aslot * DFF + aq * 8;
    unsigned alds[2];
    alds[0] = lds_lo32(&(ap ? Al : Ah)[0][ar * LDT + aq * 8]);
    alds[1] = lds_lo32(&(ap ? Al : Ah)[1][ar * LDT + aq * 8]);

    // B: 4 chunks/thread from W2t planes
    const unsigned short* bsrc[4];
    unsigned blds[4][2];
#pragma unroll
    for (int s = 0; s < 4; ++s) {
        int cid = tid + s * 256;
        int pb = cid >> 9, t3 = cid & 511, cc = t3 >> 2, qb = t3 & 3;
        bsrc[s] = (pb ? W2t_lo : W2t_hi) + (size_t)(e * DM + n0 + cc) * DFF + qb * 8;
        blds[s][0] = lds_lo32(&(pb ? Bl : Bh)[0][cc * LDT + qb * 8]);
        blds[s][1] = lds_lo32(&(pb ? Bl : Bh)[1][cc * LDT + qb * 8]);
    }

    const int mf = wv & 1, nh = wv >> 1;
    v8f acc0 = {}, acc1 = {};

    async_cp16(alds[0], asrc);
#pragma unroll
    for (int s = 0; s < 4; ++s) async_cp16(blds[s][0], bsrc[s]);

    const int NT = DFF / 32;
    for (int kt = 0; kt < NT; ++kt) {
        const int buf = kt & 1;
        if (kt + 1 < NT) {
            const int kb = (kt + 1) * 32;
            async_cp16(alds[buf ^ 1], asrc + kb);
#pragma unroll
            for (int s = 0; s < 4; ++s) async_cp16(blds[s][buf ^ 1], bsrc[s] + kb);
            WAIT_ASYNC_5();
        } else {
            WAIT_ASYNC_0();
        }
        __syncthreads();

        const unsigned short* Ahb = Ah[buf];
        const unsigned short* Alb = Al[buf];
        const unsigned short* Bhb = Bh[buf];
        const unsigned short* Blb = Bl[buf];

        v16bf ahf = load_frag_a(&Ahb[(mf * 16 + lid) * LDT], hs);
        v16bf alf = load_frag_a(&Alb[(mf * 16 + lid) * LDT], hs);
        int cc0 = nh * 32 + lid;
        int cc1 = nh * 32 + 16 + lid;
        v16bf b0h = load_frag_b(&Bhb[cc0 * LDT], hs);
        v16bf b0l = load_frag_b(&Blb[cc0 * LDT], hs);
        v16bf b1h = load_frag_b(&Bhb[cc1 * LDT], hs);
        v16bf b1l = load_frag_b(&Blb[cc1 * LDT], hs);

        acc0 = wmma_bf16(alf, b0h, acc0);
        acc0 = wmma_bf16(ahf, b0l, acc0);
        acc0 = wmma_bf16(ahf, b0h, acc0);
        acc1 = wmma_bf16(alf, b1h, acc1);
        acc1 = wmma_bf16(ahf, b1l, acc1);
        acc1 = wmma_bf16(ahf, b1h, acc1);
        __syncthreads();
    }

    const int col0 = n0 + nh * 32 + lid;
    const int col1 = col0 + 16;
    const float bb0 = b2[e * DM + col0];
    const float bb1 = b2[e * DM + col1];
#pragma unroll
    for (int j = 0; j < 8; ++j) {
        int grow = m0 + mf * 16 + j + 8 * hs;
        if (grow < c) {
            int slot  = slot_of[e * T_TOK + grow];
            float wgt = wgt_of[e * T_TOK + grow];
            yslot[(size_t)slot * DM + col0] = (acc0[j] + bb0) * wgt;
            yslot[(size_t)slot * DM + col1] = (acc1[j] + bb1) * wgt;
        }
    }
}

// ---------------------------------------------------------------------------
// combine the two expert contributions per token (slot-indexed, no atomics)
// ---------------------------------------------------------------------------
__global__ __launch_bounds__(256) void moe_combine(
    const float* __restrict__ yslot, float* __restrict__ out)
{
    size_t i = (size_t)blockIdx.x * 256 + threadIdx.x;
    size_t t = i >> 10;
    size_t cc = i & 1023;
    out[i] = yslot[(2 * t) * DM + cc] + yslot[(2 * t + 1) * DM + cc];
}

// ---------------------------------------------------------------------------
extern "C" void kernel_launch(void* const* d_in, const int* in_sizes, int n_in,
                              void* d_out, int out_size, void* d_ws, size_t ws_size,
                              hipStream_t stream) {
    const float* x    = (const float*)d_in[0];
    const float* Wr   = (const float*)d_in[1];
    const float* temp = (const float*)d_in[2];
    const float* W1   = (const float*)d_in[3];
    const float* b1   = (const float*)d_in[4];
    const float* W2   = (const float*)d_in[5];
    const float* b2   = (const float*)d_in[6];
    float* out = (float*)d_out;

    char* ws = (char*)d_ws;
    const size_t MB = 1ull << 20;
    int*            cnt     = (int*)(ws);
    int*            tok_of  = (int*)(ws + 1024);
    int*            slot_of = (int*)(ws + 1024 + 131072);
    float*          wgt_of  = (float*)(ws + 1024 + 2 * 131072);
    unsigned short* xhi     = (unsigned short*)(ws + 1 * MB);    //  8 MB
    unsigned short* xlo     = (unsigned short*)(ws + 9 * MB);    //  8 MB
    unsigned short* W1t_hi  = (unsigned short*)(ws + 17 * MB);   // 64 MB
    unsigned short* W1t_lo  = (unsigned short*)(ws + 81 * MB);   // 64 MB
    unsigned short* W2t_hi  = (unsigned short*)(ws + 145 * MB);  // 32 MB
    unsigned short* W2t_lo  = (unsigned short*)(ws + 177 * MB);  // 32 MB
    unsigned short* u_hi    = (unsigned short*)(ws + 209 * MB);  // 32 MB
    unsigned short* u_lo    = (unsigned short*)(ws + 241 * MB);  // 32 MB
    float*          yslot   = (float*)(ws + 273 * MB);           // 32 MB

    hipMemsetAsync(cnt, 0, 1024, stream);

    // one-time precision split / transpose of operands
    split_plane<<<(T_TOK * DM) / 256, 256, 0, stream>>>(x, xhi, xlo, T_TOK * DM);
    transpose_split<<<dim3(H2 / 32, DM / 32, NE), 256, 0, stream>>>(W1, W1t_hi, W1t_lo, DM, H2);
    transpose_split<<<dim3(DM / 32, DFF / 32, NE), 256, 0, stream>>>(W2, W2t_hi, W2t_lo, DFF, DM);

    moe_router <<<T_TOK / 8, 256, 0, stream>>>(x, Wr, temp, cnt, tok_of, slot_of, wgt_of);
    moe_gemm1  <<<dim3(DFF / 64, T_TOK / 32, NE), 256, 0, stream>>>(
        xhi, xlo, W1t_hi, W1t_lo, b1, cnt, tok_of, slot_of, u_hi, u_lo);
    moe_gemm2  <<<dim3(DM / 128, T_TOK / 32, NE), 256, 0, stream>>>(
        u_hi, u_lo, W2t_hi, W2t_lo, b2, cnt, slot_of, wgt_of, yslot);
    moe_combine<<<(T_TOK * DM) / 256, 256, 0, stream>>>(yslot, out);
}
